// E70MatrixLinearCell_39152921870805
// MI455X (gfx1250) — compile-verified, matrix operand order checked by reading, not run on hardware
//
#include <hip/hip_runtime.h>
#include <math.h>

#define EPSF 1e-8f

typedef _Float16 v16h __attribute__((ext_vector_type(16)));
typedef _Float16 v8h  __attribute__((ext_vector_type(8)));
typedef _Float16 v4h  __attribute__((ext_vector_type(4)));
typedef float    v8f  __attribute__((ext_vector_type(8)));

// ---------------------------------------------------------------------------
// Phase 1: spectral normalization scale factors.
// sigma(W) via 3 power iterations; emits scale = 0.5/(sigma+eps).
// One block per matrix (block 0 -> W_k/u_k, block 1 -> W_v/u_v).
// ---------------------------------------------------------------------------
__global__ __launch_bounds__(256) void spectral_kernel(
    const float* __restrict__ Wk, const float* __restrict__ Wv,
    const float* __restrict__ uk, const float* __restrict__ uv,
    float* __restrict__ scales)
{
    const float* W  = (blockIdx.x == 0) ? Wk : Wv;
    const float* u0 = (blockIdx.x == 0) ? uk : uv;
    __shared__ float u[64];
    __shared__ float uraw[64];
    __shared__ float v[1024];
    __shared__ float red[256];
    __shared__ float nrm;

    const int tid = threadIdx.x;
    if (tid < 64) u[tid] = u0[tid];
    __syncthreads();

    for (int it = 0; it < 3; ++it) {
        // v_raw[d] = sum_n W[n,d] * u[n]; each thread owns 4 columns.
        float a0 = 0.f, a1 = 0.f, a2 = 0.f, a3 = 0.f;
        for (int nr = 0; nr < 64; ++nr) {
            const float un = u[nr];
            const float* wr = W + nr * 1024 + tid * 4;
            a0 += wr[0] * un; a1 += wr[1] * un;
            a2 += wr[2] * un; a3 += wr[3] * un;
        }
        red[tid] = a0 * a0 + a1 * a1 + a2 * a2 + a3 * a3;
        __syncthreads();
        for (int off = 128; off > 0; off >>= 1) {
            if (tid < off) red[tid] += red[tid + off];
            __syncthreads();
        }
        if (tid == 0) nrm = sqrtf(red[0]) + EPSF;
        __syncthreads();
        const float inv = 1.0f / nrm;
        v[tid * 4 + 0] = a0 * inv; v[tid * 4 + 1] = a1 * inv;
        v[tid * 4 + 2] = a2 * inv; v[tid * 4 + 3] = a3 * inv;
        __syncthreads();

        // u_raw[n] = sum_d W[n,d] * v[d]; 4 threads per row n.
        const int nn = tid >> 2, part = tid & 3;
        float acc = 0.f;
        {
            const float* wr = W + nn * 1024 + part * 256;
            const float* vp = v + part * 256;
            for (int d = 0; d < 256; ++d) acc += wr[d] * vp[d];
        }
        red[tid] = acc;
        __syncthreads();
        if (part == 0)
            uraw[nn] = red[tid] + red[tid + 1] + red[tid + 2] + red[tid + 3];
        __syncthreads();
        if (tid == 0) {
            float ss = 0.f;
            for (int nr = 0; nr < 64; ++nr) ss += uraw[nr] * uraw[nr];
            nrm = sqrtf(ss) + EPSF;
        }
        __syncthreads();
        if (tid < 64) u[tid] = uraw[tid] / nrm;
        __syncthreads();
    }

    if (tid == 0) {
        // sigma = |u . (W v)| = |u . uraw|
        float sg = 0.f;
        for (int nr = 0; nr < 64; ++nr) sg += u[nr] * uraw[nr];
        sg = fabsf(sg);
        scales[blockIdx.x] = 0.5f / (sg + EPSF);
    }
}

// ---------------------------------------------------------------------------
// Phase 2a: one-time f32 -> f16 conversion of the concatenated weights.
// Wh rows: [0,64)=W_k, [64,128)=W_v, [128,192)=W_q  (each row is 1024 f16).
// ---------------------------------------------------------------------------
__global__ __launch_bounds__(256) void convert_w(
    const float* __restrict__ Wk, const float* __restrict__ Wv,
    const float* __restrict__ Wq, _Float16* __restrict__ Wh)
{
    const int e = (blockIdx.x * 256 + threadIdx.x) * 4;   // 192*1024 = 196608 elems
    const float* src;
    if (e < 65536)       src = Wk + e;
    else if (e < 131072) src = Wv + (e - 65536);
    else                 src = Wq + (e - 131072);
    const float4 f = *(const float4*)src;
    v4h h;
    h[0] = (_Float16)f.x; h[1] = (_Float16)f.y;
    h[2] = (_Float16)f.z; h[3] = (_Float16)f.w;
    *(v4h*)(Wh + e) = h;
}

// ---------------------------------------------------------------------------
// Phase 2b: one-time f32 -> f16 conversion of x  (16384 x 1024).
// ---------------------------------------------------------------------------
__global__ __launch_bounds__(256) void convert_x(
    const float* __restrict__ x, _Float16* __restrict__ xh)
{
    const size_t e = ((size_t)blockIdx.x * 256 + threadIdx.x) * 8;  // 16M elems / 8
    const float4 f0 = *(const float4*)(x + e);
    const float4 f1 = *(const float4*)(x + e + 4);
    v8h h;
    h[0] = (_Float16)f0.x; h[1] = (_Float16)f0.y;
    h[2] = (_Float16)f0.z; h[3] = (_Float16)f0.w;
    h[4] = (_Float16)f1.x; h[5] = (_Float16)f1.y;
    h[6] = (_Float16)f1.z; h[7] = (_Float16)f1.w;
    *(v8h*)(xh + e) = h;
}

// ---------------------------------------------------------------------------
// Phase 2c: projections K/V/Q = x @ W^T via WMMA f16 (f32 accumulate).
// One wave computes the SAME 16x16 tile position for K, V and Q (shared A
// operand, 3 WMMAs per K-step). 4096 wave-tiles, 4 waves/block.
// ---------------------------------------------------------------------------
__global__ __launch_bounds__(128) void proj_gemm(
    const _Float16* __restrict__ xh, const _Float16* __restrict__ Wh,
    const float* __restrict__ scales,
    float* __restrict__ Kp, float* __restrict__ Vp, float* __restrict__ Qp)
{
    const int lane = threadIdx.x & 31;
    const int wave = threadIdx.x >> 5;
    const int tile = blockIdx.x * 4 + wave;        // 0..4095
    const int mt = tile & 1023;                     // 1024 M tiles
    const int ct = tile >> 10;                      // 0..3 column tiles of n=64

    const int ncol = ct * 16 + (lane & 15);         // column within n=64
    const int mrow = mt * 16 + (lane & 15);         // global m (row of x)
    const int c    = (lane & 16) ? 8 : 0;           // K sub-offset per ISA layout

    const _Float16* arow = xh + (size_t)mrow * 1024;
    const _Float16* bk   = Wh + (size_t)(ncol)       * 1024;
    const _Float16* bv   = Wh + (size_t)(64  + ncol) * 1024;
    const _Float16* bq   = Wh + (size_t)(128 + ncol) * 1024;

    v8f accK = {}, accV = {}, accQ = {};
    for (int kb = 0; kb < 1024; kb += 32) {
        const v8h alo = *(const v8h*)(arow + kb + c);
        const v8h ahi = *(const v8h*)(arow + kb + c + 16);
        const v8h klo = *(const v8h*)(bk + kb + c);
        const v8h khi = *(const v8h*)(bk + kb + c + 16);
        const v8h vlo = *(const v8h*)(bv + kb + c);
        const v8h vhi = *(const v8h*)(bv + kb + c + 16);
        const v8h qlo = *(const v8h*)(bq + kb + c);
        const v8h qhi = *(const v8h*)(bq + kb + c + 16);
        const v16h a  = __builtin_shufflevector(alo, ahi, 0,1,2,3,4,5,6,7,8,9,10,11,12,13,14,15);
        const v16h bK = __builtin_shufflevector(klo, khi, 0,1,2,3,4,5,6,7,8,9,10,11,12,13,14,15);
        const v16h bV = __builtin_shufflevector(vlo, vhi, 0,1,2,3,4,5,6,7,8,9,10,11,12,13,14,15);
        const v16h bQ = __builtin_shufflevector(qlo, qhi, 0,1,2,3,4,5,6,7,8,9,10,11,12,13,14,15);
        accK = __builtin_amdgcn_wmma_f32_16x16x32_f16(false, a, false, bK, (short)0, accK, false, false);
        accV = __builtin_amdgcn_wmma_f32_16x16x32_f16(false, a, false, bV, (short)0, accV, false, false);
        accQ = __builtin_amdgcn_wmma_f32_16x16x32_f16(false, a, false, bQ, (short)0, accQ, false, false);
    }

    const float sk = scales[0];
    const float sv = scales[1];
    // C/D layout: lane holds N=lane%16; VGPR r holds M = r + ((lane>=16)?8:0)
    const int mbase = mt * 16 + ((lane & 16) ? 8 : 0);
#pragma unroll
    for (int r = 0; r < 8; ++r) {
        const size_t o = (size_t)(mbase + r) * 64 + ncol;
        Kp[o] = accK[r] * sk;
        Vp[o] = accV[r] * sv;
        Qp[o] = accQ[r];
    }
}

// ---------------------------------------------------------------------------
// Phase 3: sequential recurrence. 32 blocks = 8 batches x 4 row-groups (16
// rows each). Each thread owns 4 state columns (float4) in registers; streams
// S_all out with b128 stores; shfl reduction (16-lane groups) for S.q.
// ---------------------------------------------------------------------------
__global__ __launch_bounds__(256) void recurrence_kernel(
    const float* __restrict__ S0,
    const float* __restrict__ Kp, const float* __restrict__ Vp,
    const float* __restrict__ Qp,
    float* __restrict__ out, float* __restrict__ S_all)
{
    const float decay = 0.9f;
    const int b  = blockIdx.x >> 2;
    const int i  = ((blockIdx.x & 3) * 16) + (threadIdx.x >> 4);  // row 0..63
    const int jq = threadIdx.x & 15;
    const int j0 = jq * 4;

    // init state from S0 and emit S_all[0]
    float4 s = *(const float4*)(S0 + ((size_t)b * 64 + i) * 64 + j0);
    *(float4*)(S_all + ((size_t)b * 64 + i) * 64 + j0) = s;

    for (int t = 0; t < 2048; ++t) {
        const size_t m = (size_t)t * 8 + b;
        const float4 k4 = *(const float4*)(Kp + m * 64 + j0);
        const float4 q4 = *(const float4*)(Qp + m * 64 + j0);
        const float  vi = Vp[m * 64 + i];

        s.x = decay * s.x + vi * k4.x;
        s.y = decay * s.y + vi * k4.y;
        s.z = decay * s.z + vi * k4.z;
        s.w = decay * s.w + vi * k4.w;

        *(float4*)(S_all + (((size_t)(t + 1) * 8 + b) * 64 + i) * 64 + j0) = s;

        float p = s.x * q4.x + s.y * q4.y + s.z * q4.z + s.w * q4.w;
        p += __shfl_xor(p, 1, 32);
        p += __shfl_xor(p, 2, 32);
        p += __shfl_xor(p, 4, 32);
        p += __shfl_xor(p, 8, 32);
        if (jq == 0) {
            const float sq = p;
            const float o = sq * (sq / (1.0f + __expf(-sq)));  // sq * silu(sq)
            out[m * 64 + i] = o;
        }
    }
}

// ---------------------------------------------------------------------------
extern "C" void kernel_launch(void* const* d_in, const int* in_sizes, int n_in,
                              void* d_out, int out_size, void* d_ws, size_t ws_size,
                              hipStream_t stream) {
    (void)in_sizes; (void)n_in; (void)out_size; (void)ws_size;
    const float* x  = (const float*)d_in[0];
    const float* S0 = (const float*)d_in[1];
    const float* Wk = (const float*)d_in[2];
    const float* Wv = (const float*)d_in[3];
    const float* Wq = (const float*)d_in[4];
    const float* uk = (const float*)d_in[5];
    const float* uv = (const float*)d_in[6];

    float* out   = (float*)d_out;                       // [2048, 8, 64]
    float* S_all = out + (size_t)2048 * 8 * 64;         // [2049, 8, 64, 64]

    char*  ws     = (char*)d_ws;
    float* scales = (float*)ws;                         // 2 floats
    float* Kp     = (float*)(ws + 256);                 // [16384, 64] f32
    float* Vp     = Kp + (size_t)16384 * 64;
    float* Qp     = Vp + (size_t)16384 * 64;
    _Float16* Wh  = (_Float16*)(Qp + (size_t)16384 * 64);   // [192, 1024] f16
    _Float16* xh  = Wh + (size_t)192 * 1024;                // [16384, 1024] f16

    spectral_kernel<<<2, 256, 0, stream>>>(Wk, Wv, uk, uv, scales);
    convert_w<<<192, 256, 0, stream>>>(Wk, Wv, Wq, Wh);
    convert_x<<<8192, 256, 0, stream>>>(x, xh);
    proj_gemm<<<1024, 128, 0, stream>>>(xh, Wh, scales, Kp, Vp, Qp);
    recurrence_kernel<<<32, 256, 0, stream>>>(S0, Kp, Vp, Qp, out, S_all);
}
